// FormationDynamicsLayer_15925738734056
// MI455X (gfx1250) — matrix operand order repaired and checked by reading, not compiled, not run on hardware
//
#include <hip/hip_runtime.h>
#include <math.h>

typedef __attribute__((ext_vector_type(16))) _Float16 v16h;
typedef __attribute__((ext_vector_type(8)))  _Float16 v8h;
typedef __attribute__((ext_vector_type(8)))  float    v8f;

#define N_OSC   60
#define NPAD    64
#define SSTR    72                  // staging stride (halfs): 144B = 9*16B -> aligned + bank-conflict-free
#define STEPS   10
#define WPB     4                   // waves per block
#define TPB     (WPB * 32)
#define BPW     16                  // batches per wave (WMMA N now)
#define BPB     (WPB * BPW)

__global__ __launch_bounds__(TPB)
void kuramoto_wmma_15925738734056(const float* __restrict__ theta_in,
                                  const float* __restrict__ Kmat,
                                  const float* __restrict__ omega,
                                  const float* __restrict__ p_Kglobal,
                                  const float* __restrict__ p_cmod,
                                  const float* __restrict__ p_gate,
                                  const float* __restrict__ p_csf,
                                  const float* __restrict__ p_dsneg,
                                  const float* __restrict__ p_relax,
                                  float* __restrict__ theta_out,
                                  float* __restrict__ coh_out)
{
    __shared__ _Float16 ldsK[NPAD * NPAD];          // K_eff padded, row-major [i][j], 8KB
    __shared__ _Float16 ldsS[WPB][BPW * SSTR];      // per-wave sin staging [b][j]
    __shared__ _Float16 ldsC[WPB][BPW * SSTR];      // per-wave cos staging [b][j]

    const int lane = threadIdx.x & 31;
    const int wave = threadIdx.x >> 5;
    const int hi16 = lane >> 4;                      // 0: lanes 0-15, 1: lanes 16-31
    const int col  = lane & 15;                      // local batch index (WMMA N); also A's M row

    const float TWO_PI     = 6.28318530717958647f;
    const float INV_TWO_PI = 0.159154943091895336f;

    // derived scalars (uniform -> scalar loads)
    const float cmod     = *p_cmod;
    const float cfac     = 1.0f / (1.0f + (*p_csf) * (*p_relax));
    const float eff_dt_t = 0.1f * cfac * INV_TWO_PI;           // dt in *turns*
    const float boost    = 1.0f + (*p_gate) * (*p_dsneg);
    const float scale    = (*p_Kglobal) * boost / (float)N_OSC;

    // stage K_eff = K * coupling_mod, zero-padded to 64x64 f16, row-major [i][j]
    for (int idx = threadIdx.x; idx < NPAD * NPAD; idx += TPB) {
        int i = idx >> 6, j = idx & 63;
        float v = (i < N_OSC && j < N_OSC) ? Kmat[i * N_OSC + j] * cmod : 0.0f;
        ldsK[idx] = (_Float16)v;
    }
    __syncthreads();

    // A operand fragments = K_eff tiles (M = osc i, K = osc j), resident in VGPRs.
    // 16-bit A 16x32 layout: M = lane&15; lanes 0-15 hold k in {0..7,16..23},
    // lanes 16-31 hold k in {8..15,24..31} (2 halfs per VGPR).
    v16h kfrag[4][2];
#pragma unroll
    for (int t = 0; t < 4; ++t) {
#pragma unroll
        for (int kc = 0; kc < 2; ++kc) {
            int k0 = kc * 32 + hi16 * 8;
            int k1 = k0 + 16;
            const _Float16* p = &ldsK[(16 * t + col) * NPAD];
            v8h lo = *(const v8h*)(p + k0);
            v8h hh = *(const v8h*)(p + k1);
#pragma unroll
            for (int q = 0; q < 8; ++q) { kfrag[t][kc][q] = lo[q]; kfrag[t][kc][q + 8] = hh[q]; }
        }
    }

    // theta in TURNS, D layout [i, b]: tile t, elem e -> i = e + 8*hi16 + 16t (M), b = col (N)
    const int bg = ((int)blockIdx.x * WPB + wave) * BPW;
    const int b  = bg + col;
    float th[4][8];
    float om[4][8];
#pragma unroll
    for (int t = 0; t < 4; ++t) {
#pragma unroll
        for (int e = 0; e < 8; ++e) {
            int i = e + 8 * hi16 + 16 * t;
            th[t][e] = (i < N_OSC) ? theta_in[b * N_OSC + i] * INV_TWO_PI : 0.0f;
            om[t][e] = (i < N_OSC) ? omega[i] : 0.0f;
        }
    }

    _Float16* myS = ldsS[wave];
    _Float16* myC = ldsC[wave];

    for (int s = 0; s < STEPS; ++s) {
        float sv[4][8], cv[4][8];
#pragma unroll
        for (int t = 0; t < 4; ++t) {
            v8h pS, pC;
#pragma unroll
            for (int e = 0; e < 8; ++e) {
                float x  = th[t][e];                    // turns
                float si = __builtin_amdgcn_sinf(x);    // v_sin_f32 (input in revolutions)
                float co = __builtin_amdgcn_cosf(x);
                sv[t][e] = si; cv[t][e] = co;
                pS[e] = (_Float16)si;
                pC[e] = (_Float16)co;
            }
            int j0 = 8 * hi16 + 16 * t;                 // 8 consecutive j per lane
            *(v8h*)&myS[col * SSTR + j0] = pS;          // one ds_store_b128
            *(v8h*)&myC[col * SSTR + j0] = pC;
        }
        // same-wave LDS RAW (DS is in-order per wave; wait anyway)
        asm volatile("s_wait_dscnt 0" ::: "memory");

        v8f zero = {};
        v8f accS[4], accC[4];
#pragma unroll
        for (int t = 0; t < 4; ++t) { accS[t] = zero; accC[t] = zero; }

#pragma unroll
        for (int kc = 0; kc < 2; ++kc) {
            // B operand (K x N = j x b), 16-bit 32x16 layout: n = lane&15 = local batch,
            // lanes 0-15 hold k in {0..7,16..23}, lanes 16-31 hold k in {8..15,24..31} (2 halfs per VGPR).
            int kb = kc * 32 + hi16 * 8;
            const _Float16* ps = &myS[col * SSTR + kb];
            const _Float16* pc = &myC[col * SSTR + kb];
            v8h s0 = *(const v8h*)(ps);
            v8h s1 = *(const v8h*)(ps + 16);
            v8h c0 = *(const v8h*)(pc);
            v8h c1 = *(const v8h*)(pc + 16);
            v16h bS, bC;
#pragma unroll
            for (int q = 0; q < 8; ++q) {
                bS[q] = s0[q]; bS[q + 8] = s1[q];
                bC[q] = c0[q]; bC[q + 8] = c1[q];
            }
#pragma unroll
            for (int t = 0; t < 4; ++t) {
                accS[t] = __builtin_amdgcn_wmma_f32_16x16x32_f16(
                    false, kfrag[t][kc], false, bS, (short)0, accS[t], false, false);
                accC[t] = __builtin_amdgcn_wmma_f32_16x16x32_f16(
                    false, kfrag[t][kc], false, bC, (short)0, accC[t], false, false);
            }
        }

        // elementwise update in turns + wrap to [-0.5, 0.5]
#pragma unroll
        for (int t = 0; t < 4; ++t) {
#pragma unroll
            for (int e = 0; e < 8; ++e) {
                float coupling = cv[t][e] * accS[t][e] - sv[t][e] * accC[t][e];
                float nth = th[t][e] + eff_dt_t * (om[t][e] + scale * coupling);
                nth = nth - rintf(nth);                 // wrap (turns)
                th[t][e] = nth;
            }
        }
    }

    // write final theta (radians); per lane: 8 consecutive osc per tile -> coalesces
#pragma unroll
    for (int t = 0; t < 4; ++t) {
#pragma unroll
        for (int e = 0; e < 8; ++e) {
            int i = e + 8 * hi16 + 16 * t;
            if (i < N_OSC) theta_out[b * N_OSC + i] = TWO_PI * th[t][e];
        }
    }

    // coherence: mean over oscillators = in-lane sum + one cross-half shuffle
    float cs = 0.0f, ss = 0.0f;
#pragma unroll
    for (int t = 0; t < 4; ++t) {
#pragma unroll
        for (int e = 0; e < 8; ++e) {
            int i = e + 8 * hi16 + 16 * t;
            float w = (i < N_OSC) ? 1.0f : 0.0f;
            cs += w * __builtin_amdgcn_cosf(th[t][e]);
            ss += w * __builtin_amdgcn_sinf(th[t][e]);
        }
    }
    cs += __shfl_xor(cs, 16, 32);
    ss += __shfl_xor(ss, 16, 32);
    if (hi16 == 0) {
        float cm = cs * (1.0f / (float)N_OSC);
        float sm = ss * (1.0f / (float)N_OSC);
        coh_out[b] = sqrtf(cm * cm + sm * sm);
    }
}

extern "C" void kernel_launch(void* const* d_in, const int* in_sizes, int n_in,
                              void* d_out, int out_size, void* d_ws, size_t ws_size,
                              hipStream_t stream) {
    (void)n_in; (void)out_size; (void)d_ws; (void)ws_size;
    const float* theta = (const float*)d_in[0];
    const float* K     = (const float*)d_in[1];
    const float* omg   = (const float*)d_in[2];
    const float* Kg    = (const float*)d_in[3];
    const float* cmod  = (const float*)d_in[4];
    const float* gate  = (const float*)d_in[5];
    const float* csf   = (const float*)d_in[6];
    const float* dsn   = (const float*)d_in[7];
    const float* rel   = (const float*)d_in[8];

    const int B = in_sizes[0] / N_OSC;               // 65536
    float* theta_out = (float*)d_out;
    float* coh_out   = theta_out + (size_t)B * N_OSC;

    const int blocks = B / BPB;                      // 65536/64 = 1024
    kuramoto_wmma_15925738734056<<<blocks, TPB, 0, stream>>>(
        theta, K, omg, Kg, cmod, gate, csf, dsn, rel, theta_out, coh_out);
}
